// SchNet_29815662969124
// MI455X (gfx1250) — matrix operand order repaired and checked by reading, not compiled, not run on hardware
//
#include <hip/hip_runtime.h>
#include <hip/hip_bf16.h>

typedef __attribute__((ext_vector_type(16))) _Float16 half16;
typedef __attribute__((ext_vector_type(8)))  float    v8f;

#define NNODES 50000
#define NEDGES 800000
#define F_IN   11
#define HDIM   128
#define DEPTH  4
#define NB     300
#define WDIM   64
#define NGRAPH 2500
#define GAMMA  10.0f
#define MU_STEP (30.0f/299.0f)
#define LOG2F_ 0.69314718056f
#define L2E    1.4426950408889634f

// distance bucketing for RBF sparsity (gaussian support ~ +-1.3)
#define NBUCK    30
#define PERM_LEN (NEDGES + NBUCK*16)
#define RBF_CUT  1.35f

// packed fragment sizes (halves)
#define P1_HALVES (DEPTH*10*4*512)   // 81920
#define P2_HALVES (DEPTH*2*8*512)    // 32768

// shifted softplus via hardware v_exp/v_log (f16-pipeline accuracy)
__device__ __forceinline__ float sspf(float x) {
    float sp = (x > 15.0f) ? x : __logf(1.0f + __expf(x));
    return sp - LOG2F_;
}

__device__ __forceinline__ void atomAdd(float* p, float v) {
#if defined(__HIP_PLATFORM_AMD__)
    unsafeAtomicAdd(p, v);
#else
    atomicAdd(p, v);
#endif
}

// ---------------------------------------------------------------------------
// Pack W1 (DEPTH,300,64) and W2 (DEPTH,64,128) f32 -> f16 in WMMA B-fragment
// order. B fragment (contiguous k order used here, 32x16): lanes 0-15 hold K 0..15,
// lanes 16-31 hold K 16..31; N = lane & 15. Rows K>=300 zero-padded (this
// also makes A-side masking of the padded basis unnecessary).
// ---------------------------------------------------------------------------
__global__ void pack_weights(const float* __restrict__ w1,
                             const float* __restrict__ w2,
                             _Float16* __restrict__ p1,
                             _Float16* __restrict__ p2) {
    int t = blockIdx.x * 256 + threadIdx.x;
    if (t < P1_HALVES) {
        int i    = t & 15;
        int lane = (t >> 4) & 31;
        int nt   = (t >> 9) & 3;
        int kc   = (t >> 11) % 10;
        int l    = (t >> 11) / 10;
        int k = kc * 32 + ((lane >= 16) ? 16 : 0) + i;
        int n = nt * 16 + (lane & 15);
        float v = (k < NB) ? w1[(l * NB + k) * WDIM + n] : 0.0f;
        p1[t] = (_Float16)v;
    } else if (t < P1_HALVES + P2_HALVES) {
        int u = t - P1_HALVES;
        int i    = u & 15;
        int lane = (u >> 4) & 31;
        int nt   = (u >> 9) & 7;
        int kc   = (u >> 12) & 1;
        int l    = u >> 13;
        int k = kc * 32 + ((lane >= 16) ? 16 : 0) + i;
        int n = nt * 16 + (lane & 15);
        p2[u] = (_Float16)w2[(l * WDIM + k) * HDIM + n];
    }
}

// ---------------------------------------------------------------------------
// Counting sort of edges by distance bucket (width 1.0) so that each 16-edge
// tile has a narrow d-range -> only 2-3 of the 10 RBF K-chunks are nonzero.
// meta: [0..31]=count, [32..63]=cursor, [64..95]=base (base[NBUCK]=total)
// ---------------------------------------------------------------------------
__global__ void init_meta(int* __restrict__ meta, int* __restrict__ perm) {
    int t = blockIdx.x * blockDim.x + threadIdx.x;
    if (t < 96) meta[t] = 0;
    if (t < PERM_LEN) perm[t] = -1;
}

__global__ void count_kernel(const float* __restrict__ dist, int* __restrict__ meta) {
    int e = blockIdx.x * blockDim.x + threadIdx.x;
    if (e >= NEDGES) return;
    int b = (int)dist[e];
    b = min(max(b, 0), NBUCK - 1);
    atomicAdd(&meta[b], 1);
}

__global__ void scan_kernel(int* __restrict__ meta) {
    if (threadIdx.x == 0 && blockIdx.x == 0) {
        int acc = 0;
        for (int b = 0; b < NBUCK; ++b) {
            meta[64 + b] = acc;
            acc += (meta[b] + 15) & ~15;   // pad each bucket to tile multiple
        }
        meta[64 + NBUCK] = acc;            // total padded edge count
    }
}

__global__ void scatter_kernel(const float* __restrict__ dist,
                               int* __restrict__ meta, int* __restrict__ perm) {
    int e = blockIdx.x * blockDim.x + threadIdx.x;
    if (e >= NEDGES) return;
    int b = (int)dist[e];
    b = min(max(b, 0), NBUCK - 1);
    int pos = meta[64 + b] + atomicAdd(&meta[32 + b], 1);
    perm[pos] = e;
}

// ---------------------------------------------------------------------------
// h = x @ emb_W + emb_b   (K=11, trivial)
// ---------------------------------------------------------------------------
__global__ void embed_kernel(const float* __restrict__ x,
                             const float* __restrict__ W,
                             const float* __restrict__ b,
                             float* __restrict__ h) {
    int t = blockIdx.x * blockDim.x + threadIdx.x;
    if (t >= NNODES * HDIM) return;
    int n = t >> 7, j = t & 127;
    float acc = b[j];
#pragma unroll
    for (int k = 0; k < F_IN; ++k)
        acc += x[n * F_IN + k] * W[k * HDIM + j];
    h[t] = acc;
}

// ---------------------------------------------------------------------------
// Edge kernel for one layer: one wave per 16-edge tile (edges permuted so a
// tile shares a narrow distance range).
//   stage1: [16,~2*32]f16(rbf on the fly, live chunks only) @ [.,64] -> ssp
//   stage2: [16,64] @ [64,128] -> w ; msg = h[src]*w ; atomicAdd h_out[dst]
// ---------------------------------------------------------------------------
__global__ __launch_bounds__(256) void edge_kernel(
        const float* __restrict__ hin, float* __restrict__ hout,
        const int* __restrict__ ei, const float* __restrict__ dist,
        const int* __restrict__ perm, const int* __restrict__ meta,
        const _Float16* __restrict__ pW1l,   // layer base: 40*512 halves
        const _Float16* __restrict__ pW2l,   // layer base: 16*512 halves
        const float* __restrict__ b1l,       // 64
        const float* __restrict__ b2l) {     // 128

    __shared__ __align__(32) _Float16 ldsB1[40 * 512];      // 40 KB
    __shared__ __align__(32) _Float16 ldsB2[16 * 512];      // 16 KB
    __shared__ __align__(32) _Float16 sh1[8][16][64];       // 16 KB staging

    // cooperative fragment staging (uint4 = 8 halves)
    {
        const uint4* s1 = (const uint4*)pW1l;
        uint4* d1 = (uint4*)ldsB1;
        for (int t = threadIdx.x; t < (40 * 512) / 8; t += 256) d1[t] = s1[t];
        const uint4* s2 = (const uint4*)pW2l;
        uint4* d2 = (uint4*)ldsB2;
        for (int t = threadIdx.x; t < (16 * 512) / 8; t += 256) d2[t] = s2[t];
    }
    __syncthreads();

    int wv   = threadIdx.x >> 5;
    int lane = threadIdx.x & 31;
    int tile = blockIdx.x * 8 + wv;
    int totalPad = meta[64 + NBUCK];
    if (tile * 16 >= totalPad) return;          // wave-uniform exit
    int e0 = tile * 16;
    int m  = lane & 15;      // row within tile (A row / C column)
    int hi = lane >> 4;      // half-wave select

    int   em = perm[e0 + m];                    // -1 = padding row
    float d   = (em >= 0) ? dist[em] : 1.0e15f;
    float dmn = (em >= 0) ? d : 1.0e30f;
    float dmx = (em >= 0) ? d : -1.0e30f;
#pragma unroll
    for (int s = 1; s < 16; s <<= 1) {
        dmn = fminf(dmn, __shfl_xor(dmn, s, 32));
        dmx = fmaxf(dmx, __shfl_xor(dmx, s, 32));
    }
    const float invCW = 1.0f / (32.0f * MU_STEP);
    int kcS = max(0, (int)floorf((dmn - RBF_CUT) * invCW));
    int kcE = min(9, (int)floorf((dmx + RBF_CUT) * invCW));
    // bounds are identical across the wave; force them scalar so the chunk
    // loop is a uniform s_cbranch loop (EXEC stays all-ones around WMMA)
    kcS = __builtin_amdgcn_readfirstlane(kcS);
    kcE = __builtin_amdgcn_readfirstlane(kcE);

    // rbf(k) = exp2(A + B*k + C*k^2), C compile-time, A/B per lane
    const float Cc = -GAMMA * L2E * MU_STEP * MU_STEP;
    float Bc = 2.0f * GAMMA * L2E * MU_STEP * d;
    float Ac = -GAMMA * L2E * d * d;

    // ---- stage 1: rbf @ W1 over live chunks only ----
    v8f c1[4] = {};
    for (int kc = kcS; kc <= kcE; ++kc) {
        half16 a;
#pragma unroll
        for (int i = 0; i < 16; ++i) {
            // A fragment, contiguous k order matching packed B: K = chunk + hi*16 + i, M = lane&15
            int k = kc * 32 + hi * 16 + i;
            float fk = (float)k;
            float r  = exp2f(fmaf(fk, fmaf(fk, Cc, Bc), Ac));
            a[i] = (_Float16)r;   // K>=300 rows hit zero-padded weights
        }
#pragma unroll
        for (int nt = 0; nt < 4; ++nt) {
            half16 b = *(const half16*)&ldsB1[(kc * 4 + nt) * 512 + lane * 16];
            c1[nt] = __builtin_amdgcn_wmma_f32_16x16x32_f16(
                false, a, false, b, (short)0, c1[nt], false, false);
        }
    }

    // bias + shifted-softplus, restage C -> A layout via LDS (f16)
#pragma unroll
    for (int nt = 0; nt < 4; ++nt) {
        int col = nt * 16 + m;
        float bias = b1l[col];
#pragma unroll
        for (int v = 0; v < 8; ++v) {
            int row = hi * 8 + v;               // C layout: M = hi*8 + vgpr
            sh1[wv][row][col] = (_Float16)sspf(c1[nt][v] + bias);
        }
    }

    // ---- stage 2: t1 @ W2 ----
    v8f c2[8] = {};
#pragma unroll
    for (int kc = 0; kc < 2; ++kc) {
        half16 a;
#pragma unroll
        for (int i = 0; i < 16; ++i) {
            int k = kc * 32 + hi * 16 + i;
            a[i] = sh1[wv][m][k];
        }
#pragma unroll
        for (int nt = 0; nt < 8; ++nt) {
            half16 b = *(const half16*)&ldsB2[(kc * 8 + nt) * 512 + lane * 16];
            c2[nt] = __builtin_amdgcn_wmma_f32_16x16x32_f16(
                false, a, false, b, (short)0, c2[nt], false, false);
        }
    }

    // ---- message + scatter-add (skip padding rows) ----
    int evs[8], srcs[8], dsts[8];
#pragma unroll
    for (int v = 0; v < 8; ++v) {
        evs[v] = perm[e0 + hi * 8 + v];
        if (evs[v] >= 0) {
            srcs[v] = ei[evs[v]];            // edge_index[0]
            dsts[v] = ei[NEDGES + evs[v]];   // edge_index[1]
        } else {
            srcs[v] = 0; dsts[v] = 0;
        }
    }
#pragma unroll
    for (int nt = 0; nt < 8; ++nt) {
        int col = nt * 16 + m;
        float bias = b2l[col];
#pragma unroll
        for (int v = 0; v < 8; ++v) {
            if (evs[v] >= 0) {
                float w   = c2[nt][v] + bias;
                float msg = hin[srcs[v] * HDIM + col] * w;
                atomAdd(&hout[dsts[v] * HDIM + col], msg);
            }
        }
    }
}

// ---------------------------------------------------------------------------
// Head: out_n = ssp(h[n] @ h1_W + h1_b) @ h2_W + h2_b ; pool by batch id
// ---------------------------------------------------------------------------
__global__ __launch_bounds__(128) void head_kernel(
        const float* __restrict__ h,
        const float* __restrict__ h1W, const float* __restrict__ h1b,
        const float* __restrict__ h2W, const float* __restrict__ h2b,
        const int* __restrict__ batch, float* __restrict__ out) {
    __shared__ float sh[HDIM];
    __shared__ float red[HDIM];
    int n = blockIdx.x, j = threadIdx.x;
    sh[j] = h[n * HDIM + j];
    __syncthreads();
    float acc = h1b[j];
#pragma unroll 8
    for (int k = 0; k < HDIM; ++k)
        acc += sh[k] * h1W[k * HDIM + j];
    red[j] = sspf(acc) * h2W[j];
    __syncthreads();
    for (int s = 64; s > 0; s >>= 1) {
        if (j < s) red[j] += red[j + s];
        __syncthreads();
    }
    if (j == 0) atomAdd(&out[batch[n]], red[0] + h2b[0]);
}

__global__ void zero_kernel(float* __restrict__ p, int n) {
    int t = blockIdx.x * blockDim.x + threadIdx.x;
    if (t < n) p[t] = 0.0f;
}

// ---------------------------------------------------------------------------
extern "C" void kernel_launch(void* const* d_in, const int* in_sizes, int n_in,
                              void* d_out, int out_size, void* d_ws, size_t ws_size,
                              hipStream_t stream) {
    const float* x    = (const float*)d_in[0];
    const int*   ei   = (const int*)d_in[1];
    const float* dist = (const float*)d_in[2];
    const int*   batch= (const int*)d_in[3];
    const float* embW = (const float*)d_in[4];
    const float* embb = (const float*)d_in[5];
    const float* w1   = (const float*)d_in[6];
    const float* b1   = (const float*)d_in[7];
    const float* w2   = (const float*)d_in[8];
    const float* b2   = (const float*)d_in[9];
    const float* h1W  = (const float*)d_in[10];
    const float* h1b  = (const float*)d_in[11];
    const float* h2W  = (const float*)d_in[12];
    const float* h2b  = (const float*)d_in[13];
    float* out = (float*)d_out;

    char* ws = (char*)d_ws;
    size_t off = 0;
    int* meta = (int*)(ws + off);               off += 512;                     // 96 ints used
    int* perm = (int*)(ws + off);               off += (size_t)PERM_LEN * 4;
    _Float16* pW1 = (_Float16*)(ws + off);      off += (size_t)P1_HALVES * 2;
    _Float16* pW2 = (_Float16*)(ws + off);      off += (size_t)P2_HALVES * 2;
    off = (off + 255) & ~(size_t)255;
    float* bufA = (float*)(ws + off);           off += (size_t)NNODES * HDIM * 4;
    float* bufB = (float*)(ws + off);

    // ---- distance bucketing (counting sort) ----
    init_meta<<<(PERM_LEN + 255) / 256, 256, 0, stream>>>(meta, perm);
    count_kernel<<<(NEDGES + 255) / 256, 256, 0, stream>>>(dist, meta);
    scan_kernel<<<1, 32, 0, stream>>>(meta);
    scatter_kernel<<<(NEDGES + 255) / 256, 256, 0, stream>>>(dist, meta, perm);

    // ---- weight packing + embedding ----
    int packThreads = P1_HALVES + P2_HALVES;
    pack_weights<<<(packThreads + 255) / 256, 256, 0, stream>>>(w1, w2, pW1, pW2);
    embed_kernel<<<(NNODES * HDIM + 255) / 256, 256, 0, stream>>>(x, embW, embb, bufA);

    // ---- conv layers ----
    const int maxTiles = PERM_LEN / 16;
    const int blocks = (maxTiles + 7) / 8;
    for (int l = 0; l < DEPTH; ++l) {
        hipMemcpyAsync(bufB, bufA, (size_t)NNODES * HDIM * sizeof(float),
                       hipMemcpyDeviceToDevice, stream);
        edge_kernel<<<blocks, 256, 0, stream>>>(
            bufA, bufB, ei, dist, perm, meta,
            pW1 + (size_t)l * 40 * 512,
            pW2 + (size_t)l * 16 * 512,
            b1 + l * WDIM, b2 + l * HDIM);
        float* t = bufA; bufA = bufB; bufB = t;
    }

    // ---- head + pool ----
    zero_kernel<<<(NGRAPH + 255) / 256, 256, 0, stream>>>(out, NGRAPH);
    head_kernel<<<NNODES, 128, 0, stream>>>(bufA, h1W, h1b, h2W, h2b, batch, out);
}
